// GraphEncoder_85959475462285
// MI455X (gfx1250) — matrix operand order repaired and checked by reading, not compiled, not run on hardware
//
#include <hip/hip_runtime.h>
#include <hip/hip_bf16.h>
#include <stdint.h>

#define F_IN 128
#define HDIM 256

typedef unsigned short u16;
typedef __attribute__((ext_vector_type(16))) __bf16 v16bf;
typedef __attribute__((ext_vector_type(8)))  float  v8f;

#define WMMA_BF16(a, b, c) \
  __builtin_amdgcn_wmma_f32_16x16x32_bf16(false, (a), false, (b), (short)0, (c), false, false)

static __device__ __forceinline__ u16 f2bf_bits(float f) {
  union { __bf16 b; u16 s; } u;
  u.b = (__bf16)f;
  return u.s;
}
static __device__ __forceinline__ float bf2f(u16 s) {
  union { unsigned u; float f; } v;
  v.u = ((unsigned)s) << 16;
  return v.f;
}

// ---------- fragment loaders ----------
// A fragment: 16x32 bf16 (M x K). Lane m = lane&15 is the row.
// lanes 0-15: halfs 0..7 = K kbase+0..7,   halfs 8..15 = K kbase+16..23
// lanes 16-31: halfs 0..7 = K kbase+8..15, halfs 8..15 = K kbase+24..31

// from f32 source (input projection)
static __device__ __forceinline__ v16bf load_a_frag_f32(const float* __restrict__ base,
                                                        long row, int ld, int kbase) {
  const int lane = threadIdx.x & 31;
  const int klo  = (lane >> 4) * 8;
  const float* p = base + row * (long)ld + kbase + klo;
  v16bf a;
#pragma unroll
  for (int j = 0; j < 8; ++j) a[j] = (__bf16)p[j];
#pragma unroll
  for (int j = 0; j < 8; ++j) a[8 + j] = (__bf16)p[16 + j];
  return a;
}

// from bf16 source, no scale: pure bit copy (two b128 loads, no VALU)
static __device__ __forceinline__ v16bf load_a_frag_bf(const u16* __restrict__ base,
                                                       long row, int kbase) {
  const int lane = threadIdx.x & 31;
  const int klo  = (lane >> 4) * 8;
  const u16* p = base + row * (long)HDIM + kbase + klo;
  union { v16bf v; uint4 q[2]; } u;
  u.q[0] = *(const uint4*)p;
  u.q[1] = *(const uint4*)(p + 16);
  return u.v;
}

// from bf16 source with f32 scale (agg * inv_deg)
static __device__ __forceinline__ v16bf load_a_frag_bf_scaled(const u16* __restrict__ base,
                                                              long row, int kbase,
                                                              float scale) {
  const int lane = threadIdx.x & 31;
  const int klo  = (lane >> 4) * 8;
  const u16* p = base + row * (long)HDIM + kbase + klo;
  uint4 c0 = *(const uint4*)p;
  uint4 c1 = *(const uint4*)(p + 16);
  const unsigned* w0 = (const unsigned*)&c0;
  const unsigned* w1 = (const unsigned*)&c1;
  v16bf a;
#pragma unroll
  for (int d = 0; d < 4; ++d) {
    unsigned w = w0[d];
    a[2 * d]     = (__bf16)(bf2f((u16)(w & 0xffffu)) * scale);
    a[2 * d + 1] = (__bf16)(bf2f((u16)(w >> 16)) * scale);
  }
#pragma unroll
  for (int d = 0; d < 4; ++d) {
    unsigned w = w1[d];
    a[8 + 2 * d]     = (__bf16)(bf2f((u16)(w & 0xffffu)) * scale);
    a[8 + 2 * d + 1] = (__bf16)(bf2f((u16)(w >> 16)) * scale);
  }
  return a;
}

// B fragment: 32x16 bf16 (K x N), B[k][n] = W[col_n][k], W row-major [out,in] in bf16.
// lanes 0-15 hold K kbase+0..7,16..23 for column n=lane&15; lanes 16-31 hold K kbase+8..15,24..31.
static __device__ __forceinline__ v16bf load_b_frag_bf(const u16* __restrict__ W,
                                                       int col, int ld, int kbase) {
  const int lane = threadIdx.x & 31;
  const int kb   = (lane >> 4) * 8;
  const u16* p = W + (long)col * ld + kbase + kb;
  union { v16bf v; uint4 q[2]; } u;
  u.q[0] = *(const uint4*)p;
  u.q[1] = *(const uint4*)(p + 16);
  return u.v;
}

// ---------- small utility kernels ----------
__global__ void zero_kernel(float* __restrict__ p, long n) {
  long i = (long)blockIdx.x * blockDim.x + threadIdx.x;
  long stride = (long)gridDim.x * blockDim.x;
  for (; i < n; i += stride) p[i] = 0.0f;
}

__global__ void f32_to_bf16_kernel(const float* __restrict__ in, u16* __restrict__ out, long n) {
  long i = (long)blockIdx.x * blockDim.x + threadIdx.x;
  long stride = (long)gridDim.x * blockDim.x;
  for (; i < n; i += stride) out[i] = f2bf_bits(in[i]);
}

__global__ void deg_kernel(const int* __restrict__ ei, float* __restrict__ deg, int E) {
  int i = blockIdx.x * blockDim.x + threadIdx.x;
  if (i < E) atomicAdd(&deg[ei[E + i]], 1.0f);
}

__global__ void invdeg_kernel(float* __restrict__ deg, int N) {
  int i = blockIdx.x * blockDim.x + threadIdx.x;
  if (i < N) deg[i] = 1.0f / fmaxf(deg[i], 1.0f);
}

// one wave32 per edge; lane handles channel pairs 2l, 2l+64, 2l+128, 2l+192.
// packed bf16 atomic adds (CDNA5 GLOBAL_ATOMIC_PK_ADD_BF16): 4 atomics/lane, 128B coalesced reads.
__global__ __launch_bounds__(256) void scatter_kernel(const int* __restrict__ ei,
                                                      const u16* __restrict__ h,
                                                      u16* __restrict__ agg, int E) {
  int w = (blockIdx.x * blockDim.x + threadIdx.x) >> 5;
  int lane = threadIdx.x & 31;
  if (w >= E) return;
  long src = ei[w];
  long dst = ei[E + w];
  const u16* hs = h + src * HDIM;
  u16* ag = agg + dst * HDIM;
#pragma unroll
  for (int it = 0; it < 4; ++it) {
    int c = 2 * lane + 64 * it;
    unsigned v = *(const unsigned*)(hs + c);
    asm volatile("global_atomic_pk_add_bf16 %0, %1, off"
                 :: "v"(ag + c), "v"(v) : "memory");
  }
}

// h = relu(x @ Wp^T + bp), output bf16. Block: 8 waves, 32 rows x 256 cols.
__global__ __launch_bounds__(256) void proj_kernel(const float* __restrict__ x,
                                                   const u16* __restrict__ Wp,
                                                   const float* __restrict__ bp,
                                                   u16* __restrict__ hout, int N) {
  const int wave = threadIdx.x >> 5;
  const int lane = threadIdx.x & 31;
  const int rowBase = blockIdx.x * 32;
  const int m = lane & 15;
  const int rsel = lane >> 4;
  const long rowA0 = min(rowBase + m, N - 1);
  const long rowA1 = min(rowBase + 16 + m, N - 1);
  const int col0 = wave * 32 + m;
  const int col1 = wave * 32 + 16 + m;

  v8f acc00 = {}, acc01 = {}, acc10 = {}, acc11 = {};
#pragma unroll
  for (int ks = 0; ks < F_IN; ks += 32) {
    v16bf a0 = load_a_frag_f32(x, rowA0, F_IN, ks);
    v16bf a1 = load_a_frag_f32(x, rowA1, F_IN, ks);
    v16bf b0 = load_b_frag_bf(Wp, col0, F_IN, ks);
    v16bf b1 = load_b_frag_bf(Wp, col1, F_IN, ks);
    acc00 = WMMA_BF16(a0, b0, acc00);
    acc01 = WMMA_BF16(a0, b1, acc01);
    acc10 = WMMA_BF16(a1, b0, acc10);
    acc11 = WMMA_BF16(a1, b1, acc11);
  }
  const float bb0 = bp[col0], bb1 = bp[col1];
#pragma unroll
  for (int r = 0; r < 8; ++r) {
    int r0 = rowBase + r + 8 * rsel;
    int r1 = r0 + 16;
    if (r0 < N) {
      hout[(long)r0 * HDIM + col0] = f2bf_bits(fmaxf(acc00[r] + bb0, 0.0f));
      hout[(long)r0 * HDIM + col1] = f2bf_bits(fmaxf(acc01[r] + bb1, 0.0f));
    }
    if (r1 < N) {
      hout[(long)r1 * HDIM + col0] = f2bf_bits(fmaxf(acc10[r] + bb0, 0.0f));
      hout[(long)r1 * HDIM + col1] = f2bf_bits(fmaxf(acc11[r] + bb1, 0.0f));
    }
  }
}

// h_next = relu(LN(agg*invdeg @ Wl^T + bl + h @ Wr^T)) + h.  Fused GEMM + LayerNorm.
// Writes f32 to houtF (final layer) if non-null, else bf16 to houtB.
__global__ __launch_bounds__(256) void layer_kernel(
    const u16* __restrict__ agg, const u16* __restrict__ hprev,
    const float* __restrict__ invdeg,
    const u16* __restrict__ Wl, const float* __restrict__ bl,
    const u16* __restrict__ Wr,
    const float* __restrict__ gamma, const float* __restrict__ beta,
    u16* __restrict__ houtB, float* __restrict__ houtF, int N) {
  __shared__ float tile[32 * 260];   // stride 260 -> no LDS bank conflicts
  __shared__ float ps1[32][8];
  __shared__ float ps2[32][8];
  __shared__ float mu[32];
  __shared__ float rsd[32];

  const int wave = threadIdx.x >> 5;
  const int lane = threadIdx.x & 31;
  const int rowBase = blockIdx.x * 32;
  const int m = lane & 15;
  const int rsel = lane >> 4;
  const long rowA0 = min(rowBase + m, N - 1);
  const long rowA1 = min(rowBase + 16 + m, N - 1);
  const float sc0 = invdeg[rowA0];
  const float sc1 = invdeg[rowA1];
  const int col0 = wave * 32 + m;
  const int col1 = wave * 32 + 16 + m;

  v8f acc00 = {}, acc01 = {}, acc10 = {}, acc11 = {};
  // neighbor-aggregate half: (agg * inv_deg) @ Wl^T
#pragma unroll 2
  for (int ks = 0; ks < HDIM; ks += 32) {
    v16bf a0 = load_a_frag_bf_scaled(agg, rowA0, ks, sc0);
    v16bf a1 = load_a_frag_bf_scaled(agg, rowA1, ks, sc1);
    v16bf b0 = load_b_frag_bf(Wl, col0, HDIM, ks);
    v16bf b1 = load_b_frag_bf(Wl, col1, HDIM, ks);
    acc00 = WMMA_BF16(a0, b0, acc00);
    acc01 = WMMA_BF16(a0, b1, acc01);
    acc10 = WMMA_BF16(a1, b0, acc10);
    acc11 = WMMA_BF16(a1, b1, acc11);
  }
  // self half: h @ Wr^T (pure bf16 bit loads, no conversion)
#pragma unroll 2
  for (int ks = 0; ks < HDIM; ks += 32) {
    v16bf a0 = load_a_frag_bf(hprev, rowA0, ks);
    v16bf a1 = load_a_frag_bf(hprev, rowA1, ks);
    v16bf b0 = load_b_frag_bf(Wr, col0, HDIM, ks);
    v16bf b1 = load_b_frag_bf(Wr, col1, HDIM, ks);
    acc00 = WMMA_BF16(a0, b0, acc00);
    acc01 = WMMA_BF16(a0, b1, acc01);
    acc10 = WMMA_BF16(a1, b0, acc10);
    acc11 = WMMA_BF16(a1, b1, acc11);
  }

  // spill z = gemm + bias to LDS for row-wise LayerNorm
  const float bb0 = bl[col0], bb1 = bl[col1];
#pragma unroll
  for (int r = 0; r < 8; ++r) {
    int lr0 = r + 8 * rsel;
    int lr1 = lr0 + 16;
    tile[lr0 * 260 + col0] = acc00[r] + bb0;
    tile[lr0 * 260 + col1] = acc01[r] + bb1;
    tile[lr1 * 260 + col0] = acc10[r] + bb0;
    tile[lr1 * 260 + col1] = acc11[r] + bb1;
  }
  __syncthreads();

  // partial sums: 8 threads per row, 32 channels each
  {
    const int row = threadIdx.x >> 3;
    const int seg = threadIdx.x & 7;
    const float* tr = &tile[row * 260 + seg * 32];
    float s = 0.0f, s2 = 0.0f;
#pragma unroll
    for (int c = 0; c < 32; ++c) { float v = tr[c]; s += v; s2 += v * v; }
    ps1[row][seg] = s;
    ps2[row][seg] = s2;
  }
  __syncthreads();
  if (threadIdx.x < 32) {
    const int row = threadIdx.x;
    float s = 0.0f, s2 = 0.0f;
#pragma unroll
    for (int j = 0; j < 8; ++j) { s += ps1[row][j]; s2 += ps2[row][j]; }
    float mean = s * (1.0f / HDIM);
    float var  = s2 * (1.0f / HDIM) - mean * mean;
    mu[row]  = mean;
    rsd[row] = rsqrtf(fmaxf(var, 0.0f) + 1e-5f);
  }
  __syncthreads();

  // normalize + affine + relu + residual, write out
  {
    const int row = threadIdx.x >> 3;
    const int seg = threadIdx.x & 7;
    const int grow = rowBase + row;
    if (grow < N) {
      const float mean = mu[row], rs = rsd[row];
      const float* tr = &tile[row * 260 + seg * 32];
      const u16* hp = hprev + (long)grow * HDIM + seg * 32;
      if (houtF) {
        float* op = houtF + (long)grow * HDIM + seg * 32;
#pragma unroll
        for (int c = 0; c < 32; ++c) {
          int gc = seg * 32 + c;
          float y = (tr[c] - mean) * rs * gamma[gc] + beta[gc];
          op[c] = fmaxf(y, 0.0f) + bf2f(hp[c]);
        }
      } else {
        u16* op = houtB + (long)grow * HDIM + seg * 32;
#pragma unroll
        for (int c = 0; c < 32; ++c) {
          int gc = seg * 32 + c;
          float y = (tr[c] - mean) * rs * gamma[gc] + beta[gc];
          op[c] = f2bf_bits(fmaxf(y, 0.0f) + bf2f(hp[c]));
        }
      }
    }
  }
}

extern "C" void kernel_launch(void* const* d_in, const int* in_sizes, int n_in,
                              void* d_out, int out_size, void* d_ws, size_t ws_size,
                              hipStream_t stream) {
  const float* x     = (const float*)d_in[0];
  const int*   ei    = (const int*)d_in[1];
  const float* Wp    = (const float*)d_in[2];
  const float* bp    = (const float*)d_in[3];
  const float* Wl    = (const float*)d_in[4];
  const float* bl    = (const float*)d_in[5];
  const float* Wr    = (const float*)d_in[6];
  const float* gamma = (const float*)d_in[7];
  const float* beta  = (const float*)d_in[8];

  const int N = in_sizes[0] / F_IN;
  const int E = in_sizes[1] / 2;
  const int L = in_sizes[5] / HDIM;
  float* out = (float*)d_out;

  const size_t NH = (size_t)N * HDIM;
  u16* hA   = (u16*)d_ws;
  u16* hB   = hA + NH;
  u16* agg  = hB + NH;
  u16* WpB  = agg + NH;
  u16* WlB  = WpB + (size_t)HDIM * F_IN;
  u16* WrB  = WlB + (size_t)L * HDIM * HDIM;
  float* deg = (float*)(WrB + (size_t)L * HDIM * HDIM);

  const int rowBlocks = (N + 31) / 32;

  // weights -> bf16 (once per call; tiny)
  f32_to_bf16_kernel<<<256, 256, 0, stream>>>(Wp, WpB, (long)HDIM * F_IN);
  f32_to_bf16_kernel<<<1024, 256, 0, stream>>>(Wl, WlB, (long)L * HDIM * HDIM);
  f32_to_bf16_kernel<<<1024, 256, 0, stream>>>(Wr, WrB, (long)L * HDIM * HDIM);

  // degree -> inverse degree (in place in deg buffer)
  zero_kernel<<<512, 256, 0, stream>>>(deg, (long)N);
  deg_kernel<<<(E + 255) / 256, 256, 0, stream>>>(ei, deg, E);
  invdeg_kernel<<<(N + 255) / 256, 256, 0, stream>>>(deg, N);

  // input projection + relu -> bf16 h
  proj_kernel<<<rowBlocks, 256, 0, stream>>>(x, WpB, bp, hA, N);

  const u16* hin = hA;
  for (int l = 0; l < L; ++l) {
    // zero bf16 agg (reuse f32 zero kernel on half the element count)
    zero_kernel<<<4096, 256, 0, stream>>>((float*)agg, (long)(NH / 2));
    scatter_kernel<<<(E * 32 + 255) / 256, 256, 0, stream>>>(ei, hin, agg, E);
    const bool last = (l == L - 1);
    u16* houtB = last ? nullptr : ((l & 1) ? hA : hB);
    layer_kernel<<<rowBlocks, 256, 0, stream>>>(
        agg, hin, deg,
        WlB + (size_t)l * HDIM * HDIM, bl + (size_t)l * HDIM,
        WrB + (size_t)l * HDIM * HDIM,
        gamma + (size_t)l * HDIM, beta + (size_t)l * HDIM,
        houtB, last ? out : nullptr, N);
    hin = last ? hA /* unused afterwards */ : houtB;
  }
}
